// RWKV_Tmix_headmixer_20830591385839
// MI455X (gfx1250) — matrix operand order repaired and checked by reading, not compiled, not run on hardware
//
#include <hip/hip_runtime.h>
#include <math.h>

// ---------------------------------------------------------------- constants
#define TT 2048
#define CC 1024
#define HHEADS 16
#define KDIM 64
#define QBLK 128          // query rows per attention stage
#define NQB (TT / QBLK)   // 16 stages

// ---------------------------------------------------------------- types
typedef unsigned short bf16_t;
typedef __attribute__((ext_vector_type(16))) __bf16 v16bf;
typedef __attribute__((ext_vector_type(8)))  float  v8f;

union FragBF {
  v16bf  v;
  uint4  q[2];
  bf16_t s[16];
};

__device__ __forceinline__ bf16_t f2bf(float f) {
  unsigned int u = __float_as_uint(f);
  unsigned int r = u + 0x7fffu + ((u >> 16) & 1u);   // round-to-nearest-even
  return (bf16_t)(r >> 16);
}

// ---------------------------------------------------------------- reductions
__device__ __forceinline__ float block_reduce_sum(float v, float* red) {
  int tid = threadIdx.x;
  red[tid] = v; __syncthreads();
  for (int off = 128; off > 0; off >>= 1) {
    if (tid < off) red[tid] += red[tid + off];
    __syncthreads();
  }
  float r = red[0]; __syncthreads();
  return r;
}
__device__ __forceinline__ float block_reduce_max(float v, float* red) {
  int tid = threadIdx.x;
  red[tid] = v; __syncthreads();
  for (int off = 128; off > 0; off >>= 1) {
    if (tid < off) red[tid] = fmaxf(red[tid], red[tid + off]);
    __syncthreads();
  }
  float r = red[0]; __syncthreads();
  return r;
}

// ---------------------------------------------------------------- fp32 -> bf16 transposed
// in: [K,N] f32 row-major; out: [N,K] bf16 row-major (i.e. out[n*K+k] = in[k*N+n])
__global__ void k_cvt_bf16_T(const float* __restrict__ in, bf16_t* __restrict__ out,
                             int K, int N) {
  int i = blockIdx.x * blockDim.x + threadIdx.x;
  if (i < K * N) {
    int k = i / N, n = i - k * N;
    out[(size_t)n * K + k] = f2bf(in[i]);
  }
}

// ---------------------------------------------------------------- token shift + LoRA mixing
// one block per token, 256 threads
__global__ void k_token_prep(const float* __restrict__ x, const float* __restrict__ shift,
                             const float* __restrict__ tmx, const float* __restrict__ tmr,
                             const float* __restrict__ tmk, const float* __restrict__ tmv,
                             const float* __restrict__ w1 /*[C,96]*/,
                             const float* __restrict__ w2 /*[3,32,C]*/,
                             bf16_t* __restrict__ xq, bf16_t* __restrict__ xk,
                             bf16_t* __restrict__ xv) {
  int t = blockIdx.x, tid = threadIdx.x;
  __shared__ float xs[CC], dxp[CC], z[CC], xxx[96];
  for (int c = tid; c < CC; c += 256) {
    float xv_  = x[t * CC + c];
    float prev = (t > 0) ? x[(t - 1) * CC + c] : shift[c];
    float dx = prev - xv_;
    xs[c] = xv_; dxp[c] = dx;
    z[c] = xv_ + dx * tmx[c];
  }
  __syncthreads();
  if (tid < 96) {
    float acc = 0.f;
    for (int c = 0; c < CC; ++c) acc += z[c] * w1[c * 96 + tid];
    xxx[tid] = tanhf(acc);
  }
  __syncthreads();
  for (int c = tid; c < CC; c += 256) {
    float mq = 0.f, mk = 0.f, mv = 0.f;
    for (int d = 0; d < 32; ++d) {
      mq += xxx[d]      * w2[(0 * 32 + d) * CC + c];
      mk += xxx[32 + d] * w2[(1 * 32 + d) * CC + c];
      mv += xxx[64 + d] * w2[(2 * 32 + d) * CC + c];
    }
    xq[t * CC + c] = f2bf(xs[c] + dxp[c] * (tmr[c] + mq));
    xk[t * CC + c] = f2bf(xs[c] + dxp[c] * (tmk[c] + mk));
    xv[t * CC + c] = f2bf(xs[c] + dxp[c] * (tmv[c] + mv));
  }
}

// ---------------------------------------------------------------- per-head LoRA coefficients
// hm layout: [4][T][H]
__global__ void k_headavg(const float* __restrict__ x, const float* __restrict__ hw1 /*[C,128]*/,
                          const float* __restrict__ hw2 /*[4,32,16]*/, float* __restrict__ hm) {
  int t = blockIdx.x, tid = threadIdx.x;
  __shared__ float xs[CC], ha[128];
  for (int c = tid; c < CC; c += 256) xs[c] = x[t * CC + c];
  __syncthreads();
  if (tid < 128) {
    float acc = 0.f;
    for (int c = 0; c < CC; ++c) acc += xs[c] * hw1[c * 128 + tid];
    ha[tid] = tanhf(acc);
  }
  __syncthreads();
  if (tid < 64) {
    int n = tid >> 4, h = tid & 15;
    float acc = 0.f;
    for (int d = 0; d < 32; ++d) acc += ha[n * 32 + d] * hw2[(n * 32 + d) * HHEADS + h];
    hm[(n * TT + t) * HHEADS + h] = acc * (1.0f / (float)HHEADS);
  }
}

// ---------------------------------------------------------------- WMMA GEMM (bf16 x bf16^T -> f32)
// A [M,K] row-major bf16, Bt [N,K] row-major bf16 (transposed weight), C [M,N] f32. K == 1024.
// block = 256 thr (8 waves); block tile M=16 x N=128; grid = (M/16, N/128)
__global__ void k_gemm_bf16(const bf16_t* __restrict__ A, const bf16_t* __restrict__ Bt,
                            float* __restrict__ C, int M, int N, int K) {
  __shared__ bf16_t As[16 * 1024];
  int row0 = blockIdx.x * 16;
  int col0 = blockIdx.y * 128;
  // stage A strip (16 rows x K) into LDS via uint4
  {
    const uint4* src = (const uint4*)(A + (size_t)row0 * K);
    uint4* dst = (uint4*)As;
    int tot = (16 * K) / 8;
    for (int i = threadIdx.x; i < tot; i += 256) dst[i] = src[i];
  }
  __syncthreads();

  int lane = threadIdx.x & 31, wave = threadIdx.x >> 5;
  int half = lane >> 4, n = lane & 15;
  int colw = col0 + wave * 16;
  const bf16_t* brow = Bt + (size_t)(colw + n) * K;   // contiguous K per output column
  v8f acc = {};
  for (int kk = 0; kk < K; kk += 32) {
    FragBF Af, Bf;
    // A: row m=n; lanes 0-15 -> K kk+{0..7,16..23}; lanes 16-31 -> +8
    const bf16_t* ar = As + n * K + kk + half * 8;
    Af.q[0] = *(const uint4*)(ar);
    Af.q[1] = *(const uint4*)(ar + 16);
    // B column colw+n: K kk+{0..7,16..23} for lanes 0-15; +8 for lanes 16-31
    const bf16_t* bp = brow + kk + half * 8;
    __builtin_prefetch(bp + 128, 0, 1);
    Bf.q[0] = *(const uint4*)(bp);
    Bf.q[1] = *(const uint4*)(bp + 16);
    acc = __builtin_amdgcn_wmma_f32_16x16x32_bf16(false, Af.v, false, Bf.v,
                                                  (short)0, acc, false, false);
  }
  float* out = C + (size_t)row0 * N + colw;
#pragma unroll
  for (int r = 0; r < 8; ++r) out[(size_t)(r + half * 8) * N + n] = acc[r];
}

// ---------------------------------------------------------------- LayerNorm -> per-head bf16 [H][T][64]
__global__ void k_ln_to_heads(const float* __restrict__ in, const float* __restrict__ g,
                              const float* __restrict__ b, bf16_t* __restrict__ out, float scale) {
  int t = blockIdx.x, tid = threadIdx.x;
  __shared__ float red[256];
  float s = 0.f, s2 = 0.f;
  for (int c = tid; c < CC; c += 256) { float v = in[t * CC + c]; s += v; s2 += v * v; }
  float mean = block_reduce_sum(s, red) * (1.0f / CC);
  float ms   = block_reduce_sum(s2, red) * (1.0f / CC);
  float rst  = rsqrtf(ms - mean * mean + 1e-5f);
  for (int c = tid; c < CC; c += 256) {
    float v = (in[t * CC + c] - mean) * rst * g[c] + b[c];
    out[((size_t)(c >> 6) * TT + t) * KDIM + (c & 63)] = f2bf(v * scale);
  }
}

// ---------------------------------------------------------------- LayerNorm -> per-head TRANSPOSED bf16 [H][64][T]
__global__ void k_ln_to_heads_T(const float* __restrict__ in, const float* __restrict__ g,
                                const float* __restrict__ b, bf16_t* __restrict__ out) {
  int t = blockIdx.x, tid = threadIdx.x;
  __shared__ float red[256];
  float s = 0.f, s2 = 0.f;
  for (int c = tid; c < CC; c += 256) { float v = in[t * CC + c]; s += v; s2 += v * v; }
  float mean = block_reduce_sum(s, red) * (1.0f / CC);
  float ms   = block_reduce_sum(s2, red) * (1.0f / CC);
  float rst  = rsqrtf(ms - mean * mean + 1e-5f);
  for (int c = tid; c < CC; c += 256) {
    float v = (in[t * CC + c] - mean) * rst * g[c] + b[c];
    // out[h][kv][t], h = c>>6, kv = c&63
    out[((size_t)(c >> 6) * KDIM + (c & 63)) * TT + t] = f2bf(v);
  }
}

// ---------------------------------------------------------------- LayerNorm, row-major bf16 out [T,C]
__global__ void k_ln_rows(const float* __restrict__ in, const float* __restrict__ g,
                          const float* __restrict__ b, bf16_t* __restrict__ out) {
  int t = blockIdx.x, tid = threadIdx.x;
  __shared__ float red[256];
  float s = 0.f, s2 = 0.f;
  for (int c = tid; c < CC; c += 256) { float v = in[t * CC + c]; s += v; s2 += v * v; }
  float mean = block_reduce_sum(s, red) * (1.0f / CC);
  float ms   = block_reduce_sum(s2, red) * (1.0f / CC);
  float rst  = rsqrtf(ms - mean * mean + 1e-5f);
  for (int c = tid; c < CC; c += 256)
    out[(size_t)t * CC + c] = f2bf((in[t * CC + c] - mean) * rst * g[c] + b[c]);
}

// ---------------------------------------------------------------- attention scores (WMMA, K=64)
// q,k: bf16 [H][T][64]; yblk: f32 [H][QBLK][T]
// grid = (T/128 s-chunks, QBLK/16 tq tiles, H); block 256 (8 waves, one s-tile each)
__global__ void k_scores(const bf16_t* __restrict__ q, const bf16_t* __restrict__ k,
                         float* __restrict__ yblk, int qb) {
  __shared__ bf16_t Aq[16 * KDIM];  // 2 KB: the q tile, shared by all 8 waves
  int lane = threadIdx.x & 31, wave = threadIdx.x >> 5;
  int h = blockIdx.z;
  int tq0 = blockIdx.y * 16;            // within query block
  int tg0 = qb * QBLK + tq0;            // global query row
  int s0  = blockIdx.x * 128 + wave * 16;
  int half = lane >> 4, n = lane & 15;
  {
    const uint4* src = (const uint4*)(q + ((size_t)h * TT + tg0) * KDIM);
    uint4* dst = (uint4*)Aq;
    for (int i = threadIdx.x; i < (16 * KDIM) / 8; i += 256) dst[i] = src[i];
  }
  __syncthreads();
  const bf16_t* kh = k + (size_t)h * TT * KDIM;
  v8f acc = {};
#pragma unroll
  for (int kk = 0; kk < KDIM; kk += 32) {
    FragBF Af, Bf;
    const bf16_t* ar = Aq + n * KDIM + kk + half * 8;
    Af.q[0] = *(const uint4*)(ar);
    Af.q[1] = *(const uint4*)(ar + 16);
    const bf16_t* br = kh + (size_t)(s0 + n) * KDIM + kk + half * 8;  // K {0..7,16..23} + half*8
    Bf.q[0] = *(const uint4*)(br);
    Bf.q[1] = *(const uint4*)(br + 16);
    acc = __builtin_amdgcn_wmma_f32_16x16x32_bf16(false, Af.v, false, Bf.v,
                                                  (short)0, acc, false, false);
  }
  float* out = yblk + ((size_t)h * QBLK + tq0) * TT + s0;
#pragma unroll
  for (int r = 0; r < 8; ++r) out[(size_t)(r + half * 8) * TT + n] = acc[r];
}

// ---------------------------------------------------------------- cross-head mix + ALiBi + softmax + post-mix
// one block per query row t (within block qb); 256 threads
__global__ void k_mix_softmax(float* __restrict__ yblk, bf16_t* __restrict__ yb16,
                              const float* __restrict__ hm, int qb) {
  int t = blockIdx.x, tid = threadIdx.x;
  int tg = qb * QBLK + t;
  __shared__ float red[256];
  __shared__ float preq[HHEADS], postq[HHEADS];
  if (tid < HHEADS) {
    preq[tid]  = hm[((size_t)0 * TT + tg) * HHEADS + tid];
    postq[tid] = hm[((size_t)2 * TT + tg) * HHEADS + tid];
  }
  __syncthreads();

  // pass 1: pre-softmax cross-head mixing + causal ALiBi bias
  for (int s = tid; s < TT; s += 256) {
    float vals[HHEADS];
    float sum1 = 0.f, sum2 = 0.f;
    const float* prek_s = hm + ((size_t)1 * TT + s) * HHEADS;
#pragma unroll
    for (int h = 0; h < HHEADS; ++h) {
      float yv = yblk[((size_t)h * QBLK + t) * TT + s];
      vals[h] = yv;
      sum1 += yv * preq[h];
      sum2 += yv * prek_s[h];
    }
    float rel = (float)(tg - s);
#pragma unroll
    for (int h = 0; h < HHEADS; ++h) {
      float slope = exp2f(-0.5f * (float)(h + 1));
      float bias = (s <= tg) ? (-slope * rel) : -1e9f;
      yblk[((size_t)h * QBLK + t) * TT + s] = vals[h] + sum1 + sum2 + bias;
    }
  }
  __syncthreads();

  // pass 2: softmax over s for each head
  for (int h = 0; h < HHEADS; ++h) {
    float* row = yblk + ((size_t)h * QBLK + t) * TT;
    float m = -3.0e38f;
    for (int s = tid; s < TT; s += 256) m = fmaxf(m, row[s]);
    m = block_reduce_max(m, red);
    float sum = 0.f;
    for (int s = tid; s < TT; s += 256) { float e = __expf(row[s] - m); row[s] = e; sum += e; }
    sum = block_reduce_sum(sum, red);
    float inv = 1.0f / sum;
    for (int s = tid; s < TT; s += 256) row[s] *= inv;
    __syncthreads();
  }

  // pass 3: post-softmax cross-head mixing, convert to bf16 for AV GEMM
  for (int s = tid; s < TT; s += 256) {
    float vals[HHEADS];
    float sum1 = 0.f, sum2 = 0.f;
    const float* postk_s = hm + ((size_t)3 * TT + s) * HHEADS;
#pragma unroll
    for (int h = 0; h < HHEADS; ++h) {
      float yv = yblk[((size_t)h * QBLK + t) * TT + s];
      vals[h] = yv;
      sum1 += yv * postq[h];
      sum2 += yv * postk_s[h];
    }
#pragma unroll
    for (int h = 0; h < HHEADS; ++h)
      yb16[((size_t)h * QBLK + t) * TT + s] = f2bf(vals[h] + sum1 + sum2);
  }
}

// ---------------------------------------------------------------- AV (WMMA): att[t, h*64+kv] = sum_s y*v
// yb16 [H][QBLK][T] bf16, vT [H][64][T] bf16 (transposed), att [T][C] f32
// grid = (QBLK/16, H); block 128 (4 waves, each a 16-wide kv tile)
__global__ void k_av(const bf16_t* __restrict__ yb16, const bf16_t* __restrict__ vT,
                     float* __restrict__ att, int qb) {
  __shared__ bf16_t As[16 * TT];  // 64 KB: probability strip, shared by 4 waves
  int lane = threadIdx.x & 31, wave = threadIdx.x >> 5;
  int h = blockIdx.y;
  int t0 = blockIdx.x * 16;
  int half = lane >> 4, n = lane & 15;
  {
    const uint4* src = (const uint4*)(yb16 + ((size_t)h * QBLK + t0) * TT);
    uint4* dst = (uint4*)As;
    for (int i = threadIdx.x; i < (16 * TT) / 8; i += 128) dst[i] = src[i];
  }
  __syncthreads();
  int nc = wave * 16 + n;
  const bf16_t* brow = vT + ((size_t)h * KDIM + nc) * TT;  // contiguous s per (h, kv)
  v8f acc = {};
  for (int kk = 0; kk < TT; kk += 32) {
    FragBF Af, Bf;
    const bf16_t* ar = As + (size_t)n * TT + kk + half * 8;
    Af.q[0] = *(const uint4*)(ar);
    Af.q[1] = *(const uint4*)(ar + 16);
    const bf16_t* bp = brow + kk + half * 8;
    __builtin_prefetch(bp + 128, 0, 1);
    Bf.q[0] = *(const uint4*)(bp);
    Bf.q[1] = *(const uint4*)(bp + 16);
    acc = __builtin_amdgcn_wmma_f32_16x16x32_bf16(false, Af.v, false, Bf.v,
                                                  (short)0, acc, false, false);
  }
  float* out = att + (size_t)(qb * QBLK + t0) * CC + h * KDIM + wave * 16;
#pragma unroll
  for (int r = 0; r < 8; ++r) out[(size_t)(r + half * 8) * CC + n] = acc[r];
}

// ---------------------------------------------------------------- host launcher
extern "C" void kernel_launch(void* const* d_in, const int* in_sizes, int n_in,
                              void* d_out, int out_size, void* d_ws, size_t ws_size,
                              hipStream_t stream) {
  (void)in_sizes; (void)n_in; (void)out_size; (void)ws_size;
  const float* x     = (const float*)d_in[0];
  const float* shift = (const float*)d_in[1];
  const float* tmx   = (const float*)d_in[2];
  const float* tmr   = (const float*)d_in[3];
  const float* tmk   = (const float*)d_in[4];
  const float* tmv   = (const float*)d_in[5];
  const float* mw1   = (const float*)d_in[6];
  const float* mw2   = (const float*)d_in[7];
  const float* hw1   = (const float*)d_in[8];
  const float* hw2   = (const float*)d_in[9];
  const float* w_r   = (const float*)d_in[10];
  const float* w_k   = (const float*)d_in[11];
  const float* w_v   = (const float*)d_in[12];
  const float* w_o   = (const float*)d_in[13];
  const float* ln_r_g = (const float*)d_in[14];
  const float* ln_r_b = (const float*)d_in[15];
  const float* ln_k_g = (const float*)d_in[16];
  const float* ln_k_b = (const float*)d_in[17];
  const float* ln_v_g = (const float*)d_in[18];
  const float* ln_v_b = (const float*)d_in[19];
  const float* ln_x_g = (const float*)d_in[20];
  const float* ln_x_b = (const float*)d_in[21];
  float* out = (float*)d_out;

  char* ws = (char*)d_ws;
  size_t off = 0;
  auto alloc = [&](size_t bytes) -> void* {
    off = (off + 255) & ~(size_t)255;
    void* p = ws + off;
    off += bytes;
    return p;
  };

  bf16_t* wr16 = (bf16_t*)alloc((size_t)CC * CC * 2);   // transposed [N,K]
  bf16_t* wk16 = (bf16_t*)alloc((size_t)CC * CC * 2);
  bf16_t* wv16 = (bf16_t*)alloc((size_t)CC * CC * 2);
  bf16_t* wo16 = (bf16_t*)alloc((size_t)CC * CC * 2);
  bf16_t* xq16 = (bf16_t*)alloc((size_t)TT * CC * 2);
  bf16_t* xk16 = (bf16_t*)alloc((size_t)TT * CC * 2);
  bf16_t* xv16 = (bf16_t*)alloc((size_t)TT * CC * 2);
  float*  hm   = (float*) alloc((size_t)4 * TT * HHEADS * 4);
  float*  pre  = (float*) alloc((size_t)TT * CC * 4);
  bf16_t* q16  = (bf16_t*)alloc((size_t)TT * CC * 2);   // [H][T][64]
  bf16_t* k16  = (bf16_t*)alloc((size_t)TT * CC * 2);   // [H][T][64]
  bf16_t* vT16 = (bf16_t*)alloc((size_t)TT * CC * 2);   // [H][64][T]
  float*  yblk = (float*) alloc((size_t)HHEADS * QBLK * TT * 4);
  bf16_t* yb16 = (bf16_t*)alloc((size_t)HHEADS * QBLK * TT * 2);
  float*  att  = (float*) alloc((size_t)TT * CC * 4);
  bf16_t* att16= (bf16_t*)alloc((size_t)TT * CC * 2);

  const int nW = CC * CC;
  k_cvt_bf16_T<<<(nW + 255) / 256, 256, 0, stream>>>(w_r, wr16, CC, CC);
  k_cvt_bf16_T<<<(nW + 255) / 256, 256, 0, stream>>>(w_k, wk16, CC, CC);
  k_cvt_bf16_T<<<(nW + 255) / 256, 256, 0, stream>>>(w_v, wv16, CC, CC);
  k_cvt_bf16_T<<<(nW + 255) / 256, 256, 0, stream>>>(w_o, wo16, CC, CC);

  k_token_prep<<<TT, 256, 0, stream>>>(x, shift, tmx, tmr, tmk, tmv, mw1, mw2,
                                       xq16, xk16, xv16);
  k_headavg<<<TT, 256, 0, stream>>>(x, hw1, hw2, hm);

  dim3 ggrid(TT / 16, CC / 128);
  const float qscale = 0.125f;  // 64^-0.5

  k_gemm_bf16<<<ggrid, 256, 0, stream>>>(xq16, wr16, pre, TT, CC, CC);
  k_ln_to_heads<<<TT, 256, 0, stream>>>(pre, ln_r_g, ln_r_b, q16, qscale);
  k_gemm_bf16<<<ggrid, 256, 0, stream>>>(xk16, wk16, pre, TT, CC, CC);
  k_ln_to_heads<<<TT, 256, 0, stream>>>(pre, ln_k_g, ln_k_b, k16, 1.0f);
  k_gemm_bf16<<<ggrid, 256, 0, stream>>>(xv16, wv16, pre, TT, CC, CC);
  k_ln_to_heads_T<<<TT, 256, 0, stream>>>(pre, ln_v_g, ln_v_b, vT16);

  for (int qb = 0; qb < NQB; ++qb) {
    k_scores<<<dim3(TT / 128, QBLK / 16, HHEADS), 256, 0, stream>>>(q16, k16, yblk, qb);
    k_mix_softmax<<<QBLK, 256, 0, stream>>>(yblk, yb16, hm, qb);
    k_av<<<dim3(QBLK / 16, HHEADS), 128, 0, stream>>>(yb16, vT16, att, qb);
  }

  k_ln_rows<<<TT, 256, 0, stream>>>(att, ln_x_g, ln_x_b, att16);
  k_gemm_bf16<<<ggrid, 256, 0, stream>>>(att16, wo16, out, TT, CC, CC);
}
